// ImplicitFlowDecoder_71494025609423
// MI455X (gfx1250) — matrix operand order repaired and checked by reading, not compiled, not run on hardware
//
#include <hip/hip_runtime.h>
#include <hip/hip_bf16.h>
#include <math.h>

// ---------------------------------------------------------------------------
// MI455X (gfx1250) fused implicit-flow decoder.
// v_wmma_f32_16x16x32_f16, 32 rows (2 M-tiles) per wave32 to halve B-fragment
// L2 traffic; branch-free fast activations to keep VALU lean.
// ---------------------------------------------------------------------------

typedef __attribute__((ext_vector_type(16))) _Float16 half16;
typedef __attribute__((ext_vector_type(8)))  float    float8;
typedef __attribute__((ext_vector_type(4)))  unsigned u32x4;

#define B_   2
#define H_   384
#define W_   512
#define H8_  48
#define W8_  64
#define H16_ 24
#define W16_ 32
#define NQ_  65536
#define D_   128
#define LDY_ 132   // padded f32 LN buffer stride (breaks bank aliasing)

// Weight table: {pp_w1, pp_w2, p8_w, p16_w, f1_ph_w, f1_w1, f1_w2,
//                f2_ph_w, f2_w1, f2_w2, h_w1, h_w2}
static constexpr int WOFF[13] = {0, 6144, 14336, 22528, 30720, 38912, 55296,
                                 71680, 79872, 96256, 112640, 122880, 126976};

struct P37 { const float* p[37]; };

// Branch-free erf-based exact GELU (A&S 7.1.26, |err| < 1.5e-7).
static __device__ __forceinline__ float geluf(float x) {
  float xs = x * 0.70710678118654752f;
  float ax = fabsf(xs);
  float t  = __builtin_amdgcn_rcpf(1.0f + 0.3275911f * ax);
  float p  = ((((1.061405429f * t - 1.453152027f) * t + 1.421413741f) * t
               - 0.284496736f) * t + 0.254829592f) * t;
  float erfv = 1.0f - p * __expf(-ax * ax);
  return 0.5f * x * (1.0f + copysignf(erfv, xs));
}
static __device__ __forceinline__ float sigmf(float x) {
  return __builtin_amdgcn_rcpf(1.0f + __expf(-x));
}

static __device__ __forceinline__ void bilin4(float px, float py, int Wf, int Hf,
    int& o00, int& o01, int& o10, int& o11,
    float& w00, float& w01, float& w10, float& w11) {
  px = fminf(fmaxf(px, 0.0f), (float)(Wf - 1));
  py = fminf(fmaxf(py, 0.0f), (float)(Hf - 1));
  float xf = floorf(px), yf = floorf(py);
  int x0 = (int)xf, y0 = (int)yf;
  int x1 = (x0 + 1 < Wf) ? x0 + 1 : Wf - 1;
  int y1 = (y0 + 1 < Hf) ? y0 + 1 : Hf - 1;
  float wx = px - xf, wy = py - yf;
  o00 = y0 * Wf + x0; o01 = y0 * Wf + x1;
  o10 = y1 * Wf + x0; o11 = y1 * Wf + x1;
  w00 = (1.0f - wx) * (1.0f - wy); w01 = wx * (1.0f - wy);
  w10 = (1.0f - wx) * wy;          w11 = wx * wy;
}

// ---------------------------------------------------------------------------
// Weight prep: f32 row-major [K][N] -> f16 WMMA B-fragments.
// B (32x16 f16): lanes 0-15 hold K=kt*32+0..15 (N=lane), lanes 16-31 hold
// K=kt*32+16..31 (N=lane-16); dword v packs (K=2v, K=2v+1).
// ---------------------------------------------------------------------------
__global__ void prep_weight(const float* __restrict__ Wsrc, unsigned* __restrict__ dst,
                            int Kact, int Nact, int KT, int NT, int total) {
  int i = blockIdx.x * blockDim.x + threadIdx.x;
  if (i >= total) return;
  int v    = i & 7;
  int lane = (i >> 3) & 31;
  int t    = i >> 8;           // nt*KT + kt
  int kt = t % KT, nt = t / KT;
  int n  = nt * 16 + (lane & 15);
  int hi = lane >> 4;
  int k0 = kt * 32 + hi * 16 + 2 * v;
  float a = (k0     < Kact && n < Nact) ? Wsrc[k0 * Nact + n]       : 0.0f;
  float b = (k0 + 1 < Kact && n < Nact) ? Wsrc[(k0 + 1) * Nact + n] : 0.0f;
  union { _Float16 h[2]; unsigned u; } pk;
  pk.h[0] = (_Float16)a; pk.h[1] = (_Float16)b;
  dst[i] = pk.u;
}

// ---------------------------------------------------------------------------
// 32-row GEMM (MT=2 M-tiles): Y = act(A[32,KT*32] @ Wfrag + bias)
// MODE 0: gelu -> f16 | MODE 1: identity -> f16 | MODE 2: raw f32
// MODE 3: gated fuse -> f16:  out = fn + sigmoid(gate[n]) * acc
// ---------------------------------------------------------------------------
template <int MODE, int MT>
static __device__ __forceinline__ void run_gemm(
    const _Float16* __restrict__ sA, int ldA, int KT,
    const unsigned* __restrict__ frag, int NT,
    const float* __restrict__ bias,
    _Float16* outH, float* outF, int ldO,
    const _Float16* fnH, const float* __restrict__ gate, int lane) {
  int mlo = lane & 15, hi = lane >> 4;
  for (int nt = 0; nt < NT; ++nt) {
    float bv = bias[nt * 16 + mlo];
    float gv = 0.0f;
    if constexpr (MODE == 3) gv = sigmf(gate[nt * 16 + mlo]);
    float8 acc[MT];
#pragma unroll
    for (int mt = 0; mt < MT; ++mt) acc[mt] = {bv, bv, bv, bv, bv, bv, bv, bv};
    for (int kt = 0; kt < KT; ++kt) {
      union { u32x4 u[2]; half16 v; } b;
      const u32x4* pb = (const u32x4*)(frag + (((nt * KT + kt) * 32 + lane) << 3));
      b.u[0] = pb[0]; b.u[1] = pb[1];
#pragma unroll
      for (int mt = 0; mt < MT; ++mt) {
        union { u32x4 u[2]; half16 v; } a;
        const _Float16* ap = sA + (mt * 16 + mlo) * ldA + kt * 32 + hi * 16;
        a.u[0] = *(const u32x4*)(ap);
        a.u[1] = *(const u32x4*)(ap + 8);
        acc[mt] = __builtin_amdgcn_wmma_f32_16x16x32_f16(
            false, a.v, false, b.v, (short)0, acc[mt], false, false);
      }
    }
    int n = nt * 16 + mlo;
#pragma unroll
    for (int mt = 0; mt < MT; ++mt) {
#pragma unroll
      for (int r = 0; r < 8; ++r) {
        int mm = mt * 16 + hi * 8 + r;
        float v = acc[mt][r];
        if constexpr (MODE == 0)      outH[mm * ldO + n] = (_Float16)geluf(v);
        else if constexpr (MODE == 1) outH[mm * ldO + n] = (_Float16)v;
        else if constexpr (MODE == 2) outF[mm * ldO + n] = v;
        else {
          float f = (float)fnH[mm * 128 + n];
          outH[mm * ldO + n] = (_Float16)(f + gv * v);
        }
      }
    }
  }
}

static __device__ __forceinline__ void setup_geo(
    const float* qS, int* geoI, float* geoW, int lane, int tile, int Wf, int Hf) {
  int row = tile * 32 + lane;
  int bb  = row / NQ_;
  float px = qS[lane * 2]     * (float)(Wf - 1) / (float)(W_ - 1);
  float py = qS[lane * 2 + 1] * (float)(Hf - 1) / (float)(H_ - 1);
  int o00, o01, o10, o11; float w00, w01, w10, w11;
  bilin4(px, py, Wf, Hf, o00, o01, o10, o11, w00, w01, w10, w11);
  int base = bb * D_ * Hf * Wf;
  geoI[lane * 4 + 0] = base + o00; geoI[lane * 4 + 1] = base + o01;
  geoI[lane * 4 + 2] = base + o10; geoI[lane * 4 + 3] = base + o11;
  geoW[lane * 4 + 0] = w00; geoW[lane * 4 + 1] = w01;
  geoW[lane * 4 + 2] = w10; geoW[lane * 4 + 3] = w11;
}

static __device__ __forceinline__ void gather_feat(
    const float* __restrict__ feat, int HfWf,
    const int* geoI, const float* geoW, _Float16* X, int lane) {
  for (int t = lane; t < 32 * D_; t += 32) {
    int m = t >> 7, c = t & (D_ - 1);
    const int* gi = geoI + m * 4; const float* gw = geoW + m * 4;
    int cs = c * HfWf;
    float v = gw[0] * feat[gi[0] + cs] + gw[1] * feat[gi[1] + cs]
            + gw[2] * feat[gi[2] + cs] + gw[3] * feat[gi[3] + cs];
    X[m * D_ + c] = (_Float16)v;
  }
}

// Per-lane layernorm: lane owns one full row (32 rows == 32 lanes).
static __device__ __forceinline__ void layernorm32(
    const float* Yf, _Float16* XH, const float* __restrict__ lnw,
    const float* __restrict__ lnb, int lane) {
  float s = 0.0f, s2 = 0.0f;
  for (int c = 0; c < 128; ++c) {
    float v = Yf[lane * LDY_ + c];
    s += v; s2 += v * v;
  }
  float mean = s * (1.0f / 128.0f);
  float var  = s2 * (1.0f / 128.0f) - mean * mean;
  float rstd = rsqrtf(var + 1e-5f);
  for (int c = 0; c < 128; ++c) {
    float v = (Yf[lane * LDY_ + c] - mean) * rstd * lnw[c] + lnb[c];
    XH[lane * 128 + c] = (_Float16)v;
  }
}

// ---------------------------------------------------------------------------
// Main fused kernel: 1 wave32 per block, 32 query rows per block.
// ---------------------------------------------------------------------------
__global__ __launch_bounds__(32)
void ifd_main(P37 in, const unsigned* __restrict__ wf, float* __restrict__ out) {
  __shared__ _Float16 Xa[32 * 256];   // 16 KB
  __shared__ _Float16 Xb[32 * 256];   // 16 KB
  __shared__ _Float16 XH[32 * 128];   //  8 KB (GFB carry h)
  __shared__ float    Yf[32 * LDY_];  // ~16.5 KB (f32 pre-LN)
  __shared__ float    qS[64];
  __shared__ float    coarseS[64];
  __shared__ int      geoI[128];
  __shared__ float    geoW[128];

  const int lane = threadIdx.x;
  const int tile = blockIdx.x;

  // ---- per-row setup: query coords + coarse-flow bilinear sample ----------
  {
    int row = tile * 32 + lane;
    int bb  = row / NQ_;
    const float* q = in.p[4];
    float qx = q[row * 2 + 0], qy = q[row * 2 + 1];
    qS[lane * 2] = qx; qS[lane * 2 + 1] = qy;
    float px = qx * (float)(W8_ - 1) / (float)(W_ - 1);
    float py = qy * (float)(H8_ - 1) / (float)(H_ - 1);
    int o00, o01, o10, o11; float w00, w01, w10, w11;
    bilin4(px, py, W8_, H8_, o00, o01, o10, o11, w00, w01, w10, w11);
    const float* cf = in.p[3];
    int base = bb * 2 * H8_ * W8_, cs = H8_ * W8_;
    float c0 = w00 * cf[base + o00] + w01 * cf[base + o01]
             + w10 * cf[base + o10] + w11 * cf[base + o11];
    float c1 = w00 * cf[base + cs + o00] + w01 * cf[base + cs + o01]
             + w10 * cf[base + cs + o10] + w11 * cf[base + cs + o11];
    coarseS[lane * 2]     = c0 * 8.0f;   // * W/W8
    coarseS[lane * 2 + 1] = c1 * 8.0f;   // * H/H8
  }
  __syncthreads();

  // ---- 5x5 RGB patches -> Xa[32][96] (75 valid, pad to 96) ----------------
  {
    const float* img = in.p[0];
    for (int t = lane; t < 32 * 25; t += 32) {
      int m = t / 25, p = t % 25;
      int row = tile * 32 + m, bb = row / NQ_;
      float px = qS[m * 2]     + (float)(p % 5 - 2);
      float py = qS[m * 2 + 1] + (float)(p / 5 - 2);
      int o00, o01, o10, o11; float w00, w01, w10, w11;
      bilin4(px, py, W_, H_, o00, o01, o10, o11, w00, w01, w10, w11);
      int base = bb * 3 * H_ * W_;
#pragma unroll
      for (int c = 0; c < 3; ++c) {
        int o = base + c * H_ * W_;
        float v = w00 * img[o + o00] + w01 * img[o + o01]
                + w10 * img[o + o10] + w11 * img[o + o11];
        Xa[m * 96 + p * 3 + c] = (_Float16)v;
      }
    }
    for (int t = lane; t < 32 * 21; t += 32)
      Xa[(t / 21) * 96 + 75 + (t % 21)] = (_Float16)0.0f;
  }
  __syncthreads();

  // ---- f_local = gelu(patches@pp_w1+b1)@pp_w2+b2 --------------------------
  run_gemm<0, 2>(Xa, 96, 3, wf + WOFF[0], 8, in.p[6], Xb, nullptr, 128,
                 nullptr, nullptr, lane);
  __syncthreads();
  run_gemm<1, 2>(Xb, 128, 4, wf + WOFF[1], 8, in.p[8], XH, nullptr, 128,
                 nullptr, nullptr, lane);
  __syncthreads();

  // ================= GFB block 1 (f_next = f8) =============================
  setup_geo(qS, geoI, geoW, lane, tile, W8_, H8_);
  __syncthreads();
  gather_feat(in.p[1], H8_ * W8_, geoI, geoW, Xa, lane);
  __syncthreads();
  run_gemm<1, 2>(Xa, 128, 4, wf + WOFF[2], 8, in.p[10], Xb, nullptr, 128,
                 nullptr, nullptr, lane);                       // f8 (f16)
  __syncthreads();
  run_gemm<3, 2>(XH, 128, 4, wf + WOFF[4], 8, in.p[14], Xa, nullptr, 128,
                 Xb, in.p[15], lane);                           // fused
  __syncthreads();
  run_gemm<0, 2>(Xa, 128, 4, wf + WOFF[5], 16, in.p[17], Xb, nullptr, 256,
                 nullptr, nullptr, lane);
  __syncthreads();
  run_gemm<2, 2>(Xb, 256, 8, wf + WOFF[6], 8, in.p[19], nullptr, Yf, LDY_,
                 nullptr, nullptr, lane);
  __syncthreads();
  layernorm32(Yf, XH, in.p[20], in.p[21], lane);
  __syncthreads();

  // ================= GFB block 2 (f_next = f16 features) ===================
  setup_geo(qS, geoI, geoW, lane, tile, W16_, H16_);
  __syncthreads();
  gather_feat(in.p[2], H16_ * W16_, geoI, geoW, Xa, lane);
  __syncthreads();
  run_gemm<1, 2>(Xa, 128, 4, wf + WOFF[3], 8, in.p[12], Xb, nullptr, 128,
                 nullptr, nullptr, lane);                       // f16 proj
  __syncthreads();
  run_gemm<3, 2>(XH, 128, 4, wf + WOFF[7], 8, in.p[23], Xa, nullptr, 128,
                 Xb, in.p[24], lane);                           // fused
  __syncthreads();
  run_gemm<0, 2>(Xa, 128, 4, wf + WOFF[8], 16, in.p[26], Xb, nullptr, 256,
                 nullptr, nullptr, lane);
  __syncthreads();
  run_gemm<2, 2>(Xb, 256, 8, wf + WOFF[9], 8, in.p[28], nullptr, Yf, LDY_,
                 nullptr, nullptr, lane);
  __syncthreads();
  layernorm32(Yf, XH, in.p[29], in.p[30], lane);
  __syncthreads();

  // ---- build MI = [h(128) | pe(26) | coarse(2) | pad] -> Xa[32][160] ------
  {
    const float PI = 3.14159265358979323846f;
    for (int t = lane; t < 32 * 160; t += 32) {
      int m = t / 160, c = t % 160;
      float v;
      if (c < 128) {
        v = (float)XH[m * 128 + c];
      } else if (c < 130) {
        float q = qS[m * 2 + (c - 128)] * 0.125f;
        v = (q - rintf(q)) * 2.0f;
      } else if (c < 154) {
        int e = c - 130, d = e / 12, j = e % 12;
        float q   = qS[m * 2 + d] * 0.125f;
        float loc = (q - rintf(q)) * 2.0f;
        float ang = loc * PI * (float)(1 << (j % 6));
        v = (j < 6) ? __sinf(ang) : __cosf(ang);
      } else if (c < 156) {
        v = coarseS[m * 2 + (c - 154)];
      } else {
        v = 0.0f;
      }
      Xa[m * 160 + c] = (_Float16)v;
    }
  }
  __syncthreads();

  // ---- head: gelu(MI@h_w1+b1) -> gelu(@h_w2+b2) -> @h_w3+b3 + coarse ------
  run_gemm<0, 2>(Xa, 160, 5, wf + WOFF[10], 8, in.p[32], Xb, nullptr, 128,
                 nullptr, nullptr, lane);
  __syncthreads();
  run_gemm<0, 2>(Xb, 128, 4, wf + WOFF[11], 4, in.p[34], Xa, nullptr, 64,
                 nullptr, nullptr, lane);
  __syncthreads();
  {
    const float* w3 = in.p[35];
    const float* b3 = in.p[36];
    float o0 = b3[0], o1 = b3[1];
    for (int c = 0; c < 64; ++c) {
      float z = (float)Xa[lane * 64 + c];
      o0 += z * w3[c * 2 + 0];
      o1 += z * w3[c * 2 + 1];
    }
    int row = tile * 32 + lane;
    out[row * 2 + 0] = coarseS[lane * 2 + 0] + o0;
    out[row * 2 + 1] = coarseS[lane * 2 + 1] + o1;
  }
}

// ---------------------------------------------------------------------------
extern "C" void kernel_launch(void* const* d_in, const int* in_sizes, int n_in,
                              void* d_out, int out_size, void* d_ws, size_t ws_size,
                              hipStream_t stream) {
  (void)in_sizes; (void)out_size; (void)ws_size;
  static const int widx[12] = {5, 7, 9, 11, 13, 16, 18, 22, 25, 27, 31, 33};
  static const int KTt[12]  = {3, 4, 4, 4, 4, 4, 8, 4, 4, 8, 5, 4};
  static const int NTt[12]  = {8, 8, 8, 8, 8, 16, 8, 8, 16, 8, 8, 4};
  static const int KA [12]  = {75, 128, 128, 128, 128, 128, 256, 128, 128, 256, 156, 128};
  static const int NA [12]  = {128, 128, 128, 128, 128, 256, 128, 128, 256, 128, 128, 64};

  unsigned* ws = (unsigned*)d_ws;
  for (int i = 0; i < 12; ++i) {
    int tot = NTt[i] * KTt[i] * 256;
    prep_weight<<<(tot + 255) / 256, 256, 0, stream>>>(
        (const float*)d_in[widx[i]], ws + WOFF[i], KA[i], NA[i], KTt[i], NTt[i], tot);
  }

  P37 P;
  for (int i = 0; i < 37 && i < n_in; ++i) P.p[i] = (const float*)d_in[i];

  const int nrows  = B_ * NQ_;          // 131072
  const int blocks = nrows / 32;        // 4096
  ifd_main<<<blocks, 32, 0, stream>>>(P, (const unsigned*)d_ws, (float*)d_out);
}
